// SparseMoE_52596169507136
// MI455X (gfx1250) — compile-verified, matrix operand order checked by reading, not run on hardware
//
#include <hip/hip_runtime.h>
#include <stdint.h>

#define DIM    1024
#define HID    2048
#define NE     8
#define TILE_M 32
#define NTHR   512      // 16 waves
#define XS     1032     // sm_x row stride (u16): 2064B, 16B aligned, conflict-free
#define HS     2056     // sm_h row stride (u16)
#define WS     40       // sm_w k-stride (u16): 80B, 16B aligned, conflict-free
#define NCHUNK 512
#define SLAB   16384    // elements per pre-tiled weight slab (512n x 32k)
#define WBUFB  (NCHUNK * WS * 2)   // bytes per sm_w buffer = 40960

typedef __attribute__((ext_vector_type(16))) __bf16 v16bf;
typedef __attribute__((ext_vector_type(8)))  float  v8f;

struct FragU { uint4 lo, hi; };

static __device__ __forceinline__ unsigned short f2bf(float f) {
  union { float f; unsigned u; } v; v.f = f;
  unsigned r = v.u + 0x7FFFu + ((v.u >> 16) & 1u);
  return (unsigned short)(r >> 16);
}
static __device__ __forceinline__ float bf2f(unsigned short h) {
  union { unsigned u; float f; } v; v.u = ((unsigned)h) << 16;
  return v.f;
}
// 16-element bf16 fragment: two 16B LDS loads (K=base..+7 and K=base+16..+23)
static __device__ __forceinline__ v16bf ldfrag(const unsigned short* p) {
  FragU f;
  f.lo = *(const uint4*)(p);
  f.hi = *(const uint4*)(p + 16);
  return __builtin_bit_cast(v16bf, f);
}
// LDS byte offset of a generic pointer to __shared__ (low 32 bits of generic addr)
static __device__ __forceinline__ unsigned ldsoff(const void* p) {
  return (unsigned)(size_t)p;
}
// CDNA5 async global->LDS copy, 16B per lane, tracked by ASYNCcnt
static __device__ __forceinline__ void async_b128(unsigned lds, unsigned long long ga) {
  asm volatile("global_load_async_to_lds_b128 %0, %1, off"
               :: "v"(lds), "v"(ga) : "memory");
}
static __device__ __forceinline__ void wait_async(void) {   // all async done
  asm volatile("s_wait_asynccnt 0x0" ::: "memory");
}
static __device__ __forceinline__ void wait_async4(void) {  // all but newest slab done
  asm volatile("s_wait_asynccnt 0x4" ::: "memory");
}
// copy one pre-tiled 32KB slab (contiguous in global) into padded LDS layout
static __device__ __forceinline__ void issue_slab(const unsigned short* gslab,
                                                  unsigned lds_base, int tid) {
  unsigned long long ga = (unsigned long long)gslab + (size_t)tid * 64;  // n-row = tid
  unsigned lds = lds_base + (unsigned)tid * 80;                           // WS*2 bytes
  #pragma unroll
  for (int c = 0; c < 4; ++c) async_b128(lds + c * 16, ga + c * 16);
}

// ---------------- fp32 -> bf16 bulk convert ----------------
__global__ void moe_cvt_bf16(const float* __restrict__ in, unsigned short* __restrict__ out,
                             int n4) {
  int i = blockIdx.x * 256 + threadIdx.x;
  if (i >= n4) return;
  float4 v = ((const float4*)in)[i];
  union { unsigned short u[4]; unsigned long long ll; } p;
  p.u[0] = f2bf(v.x); p.u[1] = f2bf(v.y); p.u[2] = f2bf(v.z); p.u[3] = f2bf(v.w);
  ((unsigned long long*)out)[i] = p.ll;
}

// ------- pre-tile weights: [E][K][N] fp32 -> slabs [E][N/512][K/32][512n][32k] bf16 -------
__global__ void moe_tile_w(const float* __restrict__ w, unsigned short* __restrict__ wt,
                           int K, int N) {
  int slabsK = K >> 5, chunksN = N >> 9;
  int per_e = slabsK * chunksN;
  int b = blockIdx.x;
  int e = b / per_e, rem = b % per_e;
  int ncI = rem / slabsK, ks = rem % slabsK;
  const float* we = w + (size_t)e * K * N;
  __shared__ unsigned short sm[NCHUNK * WS];          // 40 KB
  for (int i = threadIdx.x; i < 32 * NCHUNK; i += 256) {
    int kk = i >> 9, nn = i & (NCHUNK - 1);           // coalesced read along n
    sm[nn * WS + kk] = f2bf(we[(size_t)(ks * 32 + kk) * N + ncI * NCHUNK + nn]);
  }
  __syncthreads();
  unsigned* dst = (unsigned*)(wt + ((size_t)e * per_e + (size_t)ncI * slabsK + ks) * SLAB);
  for (int i = threadIdx.x; i < SLAB / 2; i += 256) { // coalesced linear write
    int n = i >> 4, k2 = (i & 15) << 1;
    unsigned lo = sm[n * WS + k2], hi = sm[n * WS + k2 + 1];
    dst[i] = lo | (hi << 16);
  }
}

// ---------------- gating (wave per token) ----------------
__global__ void moe_gate(const float* __restrict__ x, const float* __restrict__ gw,
                         int2* __restrict__ topi, float2* __restrict__ topw, int T) {
  int wv = threadIdx.x >> 5, lane = threadIdx.x & 31;
  int t = blockIdx.x * 8 + wv;
  if (t >= T) return;
  const float* xr = x + (size_t)t * DIM;
  float acc[NE] = {};
  #pragma unroll 4
  for (int j = 0; j < DIM / 32; ++j) {
    int d = j * 32 + lane;
    float xv = xr[d];
    const float4* g4 = (const float4*)(gw + (size_t)d * NE);
    float4 a = g4[0], b = g4[1];
    acc[0] += xv * a.x; acc[1] += xv * a.y; acc[2] += xv * a.z; acc[3] += xv * a.w;
    acc[4] += xv * b.x; acc[5] += xv * b.y; acc[6] += xv * b.z; acc[7] += xv * b.w;
  }
  #pragma unroll
  for (int e = 0; e < NE; ++e)
    #pragma unroll
    for (int m = 16; m >= 1; m >>= 1) acc[e] += __shfl_xor(acc[e], m, 32);
  if (lane == 0) {
    int i1 = 0;
    #pragma unroll
    for (int e = 1; e < NE; ++e) if (acc[e] > acc[i1]) i1 = e;
    int i2 = (i1 == 0) ? 1 : 0;
    #pragma unroll
    for (int e = 0; e < NE; ++e) if (e != i1 && acc[e] > acc[i2]) i2 = e;
    float p2 = __expf(acc[i2] - acc[i1]);     // pair-renormalized softmax
    float w1 = 1.0f / (1.0f + p2);
    topi[t] = make_int2(i1, i2);
    topw[t] = make_float2(w1, 1.0f - w1);
  }
}

// ---------------- deterministic stable compaction ----------------
__global__ void moe_compact(const int2* __restrict__ topi, const float2* __restrict__ topw,
                            int* __restrict__ tok_list, float* __restrict__ slot_w,
                            int* __restrict__ counts, int T) {
  int e = blockIdx.x;
  int wv = threadIdx.x >> 5, lane = threadIdx.x & 31;
  __shared__ int wsum[8];
  __shared__ int sbase;
  if (threadIdx.x == 0) sbase = 0;
  __syncthreads();
  for (int base = 0; base < T; base += 256) {
    int t = base + threadIdx.x;
    int2   ti = topi[t];
    float2 tw = topw[t];
    bool sel = (ti.x == e) || (ti.y == e);
    unsigned mlo = (unsigned)__ballot(sel);
    int inwave = __popc(mlo & ((1u << lane) - 1u));
    if (lane == 0) wsum[wv] = __popc(mlo);
    __syncthreads();
    int woff = 0, tot = 0;
    #pragma unroll
    for (int i = 0; i < 8; ++i) { if (i < wv) woff += wsum[i]; tot += wsum[i]; }
    int myb = sbase;
    if (sel) {
      int pos = myb + woff + inwave;
      tok_list[(size_t)e * T + pos] = t;
      slot_w [(size_t)e * T + pos] = (ti.x == e) ? tw.x : tw.y;
    }
    __syncthreads();
    if (threadIdx.x == 0) sbase = myb + tot;
    __syncthreads();
  }
  if (threadIdx.x == 0) counts[e] = sbase;
}

// ---------------- fused expert FFN: bf16 WMMA + async double-buffered slabs ----------------
__global__ void moe_expert(const unsigned short* __restrict__ xbf,
                           const unsigned short* __restrict__ w1t,
                           const unsigned short* __restrict__ w2t,
                           const float* __restrict__ b1,  const float* __restrict__ g1,
                           const float* __restrict__ bt1, const float* __restrict__ b2,
                           const int* __restrict__ tok_list, const float* __restrict__ slot_w,
                           const int* __restrict__ counts, float* __restrict__ yacc,
                           int T, int maxtiles) {
  int e    = blockIdx.x / maxtiles;
  int tile = blockIdx.x % maxtiles;
  int cnt  = counts[e];
  int row0 = tile * TILE_M;
  if (row0 >= cnt) return;                         // uniform per block

  extern __shared__ __align__(16) char smem[];
  unsigned short* sm_x = (unsigned short*)smem;           // 32 x 1032
  unsigned short* sm_h = sm_x + TILE_M * XS;              // 32 x 2056
  unsigned short* sm_w = sm_h + TILE_M * HS;              // 2 x (512 x 40), double buffer
  float* sm_stats = (float*)(sm_w + 2 * NCHUNK * WS);     // 64 floats
  int*   sm_tok   = (int*)(sm_stats + 2 * TILE_M);        // 32
  float* sm_gw    = (float*)(sm_tok + TILE_M);            // 32

  int tid = threadIdx.x;
  int wv = tid >> 5, lane = tid & 31, lm = lane & 15, lh = lane >> 4;
  int mt = wv & 1;            // which 16-row M tile this wave owns
  int ng = wv >> 1;           // N-tile group (0..7), 4 tiles each

  const float* b1e  = b1  + (size_t)e * HID;
  const float* g1e  = g1  + (size_t)e * HID;
  const float* bt1e = bt1 + (size_t)e * HID;
  const float* b2e  = b2  + (size_t)e * DIM;

  if (tid < TILE_M) {
    int gr = row0 + tid;
    sm_tok[tid] = (gr < cnt) ? tok_list[(size_t)e * T + gr] : 0;
    sm_gw[tid]  = (gr < cnt) ? slot_w [(size_t)e * T + gr] : 0.0f;
  }
  if (tid < 2 * TILE_M) sm_stats[tid] = 0.0f;
  __syncthreads();

  // async gather of 32 bf16 token rows into sm_x (8 x B128 per thread)
  {
    int r = tid >> 4, seg = tid & 15;    // 16 segs of 64 elems (128B) per row
    unsigned long long ga =
        (unsigned long long)(xbf + (size_t)sm_tok[r] * DIM + seg * 64);
    unsigned lds = ldsoff(sm_x + r * XS + seg * 64);
    #pragma unroll
    for (int c = 0; c < 8; ++c) async_b128(lds + c * 16, ga + c * 16);
  }

  const unsigned smw0 = ldsoff(sm_w);

  // ---- GEMM1: h[32 x 2048] = X[32 x 1024] @ w1, 512-col chunks, K = 32 slabs ----
  for (int nc = 0; nc < HID; nc += NCHUNK) {
    const unsigned short* wbase = w1t + ((size_t)e * 128 + (size_t)(nc >> 9) * 32) * SLAB;
    issue_slab(wbase, smw0, tid);                      // slab 0 -> buf 0
    v8f acc[4] = {};
    for (int ks = 0; ks < 32; ++ks) {
      int buf = ks & 1;
      __syncthreads();                                 // everyone done reading buf^1
      if (ks + 1 < 32) { issue_slab(wbase + (size_t)(ks + 1) * SLAB,
                                    smw0 + (unsigned)(buf ^ 1) * WBUFB, tid);
                         wait_async4(); }
      else             { wait_async(); }
      __syncthreads();                                 // slab ks visible everywhere
      const unsigned short* wb = sm_w + (buf ? NCHUNK * WS : 0);
      const unsigned short* wrow = wb + lm * WS + lh * 8;
      // batch all fragment loads, then back-to-back WMMAs (single DS wait)
      v16bf A  = ldfrag(sm_x + (mt * 16 + lm) * XS + ks * 32 + lh * 8);
      v16bf B0 = ldfrag(wrow + ((ng * 4 + 0) * 16) * WS);
      v16bf B1 = ldfrag(wrow + ((ng * 4 + 1) * 16) * WS);
      v16bf B2 = ldfrag(wrow + ((ng * 4 + 2) * 16) * WS);
      v16bf B3 = ldfrag(wrow + ((ng * 4 + 3) * 16) * WS);
      acc[0] = __builtin_amdgcn_wmma_f32_16x16x32_bf16(false, A, false, B0,
                                                       (short)0, acc[0], false, false);
      acc[1] = __builtin_amdgcn_wmma_f32_16x16x32_bf16(false, A, false, B1,
                                                       (short)0, acc[1], false, false);
      acc[2] = __builtin_amdgcn_wmma_f32_16x16x32_bf16(false, A, false, B2,
                                                       (short)0, acc[2], false, false);
      acc[3] = __builtin_amdgcn_wmma_f32_16x16x32_bf16(false, A, false, B3,
                                                       (short)0, acc[3], false, false);
    }
    #pragma unroll
    for (int j = 0; j < 4; ++j) {
      int n = nc + (ng * 4 + j) * 16 + lm;
      float bb = b1e[n];
      #pragma unroll
      for (int i = 0; i < 8; ++i)
        sm_h[(mt * 16 + lh * 8 + i) * HS + n] = f2bf(acc[j][i] + bb);
    }
  }
  __syncthreads();

  // ---- LayerNorm stats + LN + exact GELU, in place on sm_h ----
  {
    int r = tid >> 4, seg = tid & 15;                  // 32 rows x 16 segs of 128
    unsigned short* hp = sm_h + r * HS + seg * 128;
    float s = 0.f, s2 = 0.f;
    #pragma unroll 8
    for (int c = 0; c < 128; ++c) { float v = bf2f(hp[c]); s += v; s2 += v * v; }
    atomicAdd(&sm_stats[r * 2 + 0], s);
    atomicAdd(&sm_stats[r * 2 + 1], s2);
    __syncthreads();
    float mu  = sm_stats[r * 2 + 0] * (1.0f / HID);
    float var = sm_stats[r * 2 + 1] * (1.0f / HID) - mu * mu;
    float rs  = rsqrtf(var + 1e-5f);
    int cb = seg * 128;
    #pragma unroll 4
    for (int c = 0; c < 128; ++c) {
      float v = bf2f(hp[c]);
      v = (v - mu) * rs * g1e[cb + c] + bt1e[cb + c];
      v = 0.5f * v * (1.0f + erff(v * 0.70710678f));
      hp[c] = f2bf(v);
    }
  }
  __syncthreads();

  // ---- GEMM2: y[32 x 1024] = h @ w2, K = 64 slabs, scaled scatter-add ----
  for (int nc = 0; nc < DIM; nc += NCHUNK) {
    const unsigned short* wbase = w2t + ((size_t)e * 128 + (size_t)(nc >> 9) * 64) * SLAB;
    issue_slab(wbase, smw0, tid);
    v8f acc[4] = {};
    for (int ks = 0; ks < 64; ++ks) {
      int buf = ks & 1;
      __syncthreads();
      if (ks + 1 < 64) { issue_slab(wbase + (size_t)(ks + 1) * SLAB,
                                    smw0 + (unsigned)(buf ^ 1) * WBUFB, tid);
                         wait_async4(); }
      else             { wait_async(); }
      __syncthreads();
      const unsigned short* wb = sm_w + (buf ? NCHUNK * WS : 0);
      const unsigned short* wrow = wb + lm * WS + lh * 8;
      v16bf A  = ldfrag(sm_h + (mt * 16 + lm) * HS + ks * 32 + lh * 8);
      v16bf B0 = ldfrag(wrow + ((ng * 4 + 0) * 16) * WS);
      v16bf B1 = ldfrag(wrow + ((ng * 4 + 1) * 16) * WS);
      v16bf B2 = ldfrag(wrow + ((ng * 4 + 2) * 16) * WS);
      v16bf B3 = ldfrag(wrow + ((ng * 4 + 3) * 16) * WS);
      acc[0] = __builtin_amdgcn_wmma_f32_16x16x32_bf16(false, A, false, B0,
                                                       (short)0, acc[0], false, false);
      acc[1] = __builtin_amdgcn_wmma_f32_16x16x32_bf16(false, A, false, B1,
                                                       (short)0, acc[1], false, false);
      acc[2] = __builtin_amdgcn_wmma_f32_16x16x32_bf16(false, A, false, B2,
                                                       (short)0, acc[2], false, false);
      acc[3] = __builtin_amdgcn_wmma_f32_16x16x32_bf16(false, A, false, B3,
                                                       (short)0, acc[3], false, false);
    }
    #pragma unroll
    for (int j = 0; j < 4; ++j) {
      int n = nc + (ng * 4 + j) * 16 + lm;
      float bb = b2e[n];
      #pragma unroll
      for (int i = 0; i < 8; ++i) {
        int m = mt * 16 + lh * 8 + i;
        float val = (acc[j][i] + bb) * sm_gw[m];
        unsafeAtomicAdd(&yacc[(size_t)sm_tok[m] * DIM + n], val);  // 2 commutative adds/elem
      }
    }
  }
}

// ---------------- residual + final LayerNorm ----------------
__global__ void moe_final(const float* __restrict__ x, const float* __restrict__ yacc,
                          const float* __restrict__ g, const float* __restrict__ b,
                          float* __restrict__ out, int T) {
  int wv = threadIdx.x >> 5, lane = threadIdx.x & 31;
  int t = blockIdx.x * 8 + wv;
  if (t >= T) return;
  const float* xr = x + (size_t)t * DIM;
  const float* yr = yacc + (size_t)t * DIM;
  float v[32];
  float s = 0.f, s2 = 0.f;
  #pragma unroll
  for (int j = 0; j < 32; ++j) {
    int c = j * 32 + lane;
    v[j] = xr[c] + yr[c];
    s += v[j]; s2 += v[j] * v[j];
  }
  #pragma unroll
  for (int m = 16; m >= 1; m >>= 1) { s += __shfl_xor(s, m, 32); s2 += __shfl_xor(s2, m, 32); }
  float mu  = s * (1.0f / DIM);
  float var = s2 * (1.0f / DIM) - mu * mu;
  float rs  = rsqrtf(var + 1e-5f);
  float* orow = out + (size_t)t * DIM;
  #pragma unroll
  for (int j = 0; j < 32; ++j) {
    int c = j * 32 + lane;
    orow[c] = (v[j] - mu) * rs * g[c] + b[c];
  }
}

extern "C" void kernel_launch(void* const* d_in, const int* in_sizes, int n_in,
                              void* d_out, int out_size, void* d_ws, size_t ws_size,
                              hipStream_t stream) {
  const float* x   = (const float*)d_in[0];
  const float* gw  = (const float*)d_in[1];
  const float* w1  = (const float*)d_in[2];
  const float* b1  = (const float*)d_in[3];
  const float* g1  = (const float*)d_in[4];
  const float* bt1 = (const float*)d_in[5];
  const float* w2  = (const float*)d_in[6];
  const float* b2  = (const float*)d_in[7];
  const float* lg  = (const float*)d_in[8];
  const float* lb  = (const float*)d_in[9];
  float* out = (float*)d_out;
  int T = in_sizes[0] / DIM;                    // 4096

  char* ws = (char*)d_ws;
  size_t o = 0;
  int*    counts   = (int*)(ws + o);            o += 256;
  int2*   topi     = (int2*)(ws + o);           o += (size_t)T * sizeof(int2);
  float2* topw     = (float2*)(ws + o);         o += (size_t)T * sizeof(float2);
  int*    tok_list = (int*)(ws + o);            o += (size_t)NE * T * sizeof(int);
  float*  slot_w   = (float*)(ws + o);          o += (size_t)NE * T * sizeof(float);
  float*  yacc     = (float*)(ws + o);          o += (size_t)T * DIM * sizeof(float);
  unsigned short* xbf = (unsigned short*)(ws + o); o += (size_t)T * DIM * 2;
  unsigned short* w1t = (unsigned short*)(ws + o); o += (size_t)NE * DIM * HID * 2;
  unsigned short* w2t = (unsigned short*)(ws + o); o += (size_t)NE * HID * DIM * 2;

  hipMemsetAsync(counts, 0, 256, stream);
  hipMemsetAsync(yacc, 0, (size_t)T * DIM * sizeof(float), stream);

  // one-time per call: bf16 convert x, pre-tile weights into slab layout
  int n4 = T * DIM / 4;
  moe_cvt_bf16<<<(n4 + 255) / 256, 256, 0, stream>>>(x, xbf, n4);
  moe_tile_w<<<NE * (HID / NCHUNK) * (DIM / 32), 256, 0, stream>>>(w1, w1t, DIM, HID);
  moe_tile_w<<<NE * (DIM / NCHUNK) * (HID / 32), 256, 0, stream>>>(w2, w2t, HID, DIM);

  moe_gate<<<T / 8, 256, 0, stream>>>(x, gw, topi, topw, T);
  moe_compact<<<NE, 256, 0, stream>>>(topi, topw, tok_list, slot_w, counts, T);

  int maxtiles = T / TILE_M;                    // 128
  size_t smem = (size_t)(TILE_M * XS + TILE_M * HS + 2 * NCHUNK * WS) * sizeof(unsigned short)
              + (2 * TILE_M) * sizeof(float) + TILE_M * sizeof(int) + TILE_M * sizeof(float);
  moe_expert<<<NE * maxtiles, NTHR, smem, stream>>>(xbf, w1t, w2t, b1, g1, bt1, b2,
                                                    tok_list, slot_w, counts, yacc,
                                                    T, maxtiles);
  moe_final<<<T / 8, 256, 0, stream>>>(x, yacc, lg, lb, out, T);
}
